// LEConvMultiEdge_87514253624018
// MI455X (gfx1250) — matrix operand order repaired and checked by reading, not compiled, not run on hardware
//
#include <hip/hip_runtime.h>

// ---------------------------------------------------------------------------
// LEConv multi-edge fused kernel for gfx1250 (CDNA5, wave32, WMMA bf16).
//   out = sigmoid( V@w1 + (deg (x) V)@w2 - agg@w3 ),  agg = einsum('bijl,bjf')
// Dominant cost: streaming A (128 MB) once; compute via v_wmma_f32_16x16x32_bf16.
// ---------------------------------------------------------------------------

typedef __attribute__((ext_vector_type(16))) __bf16 v16bf;
typedef __attribute__((ext_vector_type(8)))  __bf16 v8bf;
typedef __attribute__((ext_vector_type(8)))  float  v8f;

#define BB 2
#define NN 2048
#define FF 64
#define CC 64
#define LL 4
#define KTILE 32
#define NSTEP (NN / KTILE)          // 64 j-chunks
#define XCOLS (FF + LL*FF + LL*FF)  // 576 = 64 + 256 + 256
#define NKEP  (XCOLS / KTILE)       // 18 epilogue k-chunks

__global__ __launch_bounds__(128)
void LEConvMultiEdge_87514253624018_kernel(const float* __restrict__ V,
                                           const float* __restrict__ A,
                                           const float* __restrict__ w1,
                                           const float* __restrict__ w2,
                                           const float* __restrict__ w3,
                                           float* __restrict__ out) {
    __shared__ __bf16 shV[FF][KTILE];     // V^T tile: [f][k], 4 KB
    __shared__ __bf16 shX[16][XCOLS];     // epilogue LHS: 16 x 576, 18 KB
    __shared__ float  degbuf[LL][16];

    const int tid   = threadIdx.x;
    const int wave  = tid >> 5;           // 0..3 == edge type l
    const int lane  = tid & 31;
    const int lhalf = lane >> 4;          // 0 or 1
    const int lm    = lane & 15;
    const int b     = blockIdx.x / (NN / 16);
    const int i0    = (blockIdx.x % (NN / 16)) * 16;

    // A row pointer for this lane's output row (M = lm) and this wave's l.
    const float* aPtr = A + ((size_t)(b * NN + i0 + lm) * NN) * LL + wave;
    const int kloA = lhalf * 8;           // A-frag K groups: {0..7,16..23} / {8..15,24..31}
    const int kbB  = lhalf * 16;          // unused; B frag is read at kloA, same K groups as A

    v8f acc[4];
    #pragma unroll
    for (int t = 0; t < 4; ++t) acc[t] = (v8f){0,0,0,0,0,0,0,0};
    float dsum = 0.0f;

    // --------------------------- main loop over j ---------------------------
    for (int step = 0; step < NSTEP; ++step) {
        const int j0 = step * KTILE;

        // Stage V^T (bf16) in LDS: thread -> (f = tid&63, kq = tid>>6).
        {
            const int f  = tid & 63;
            const int kq = tid >> 6;      // 0..1, 16 k each
            const float* vp = V + (size_t)(b * NN + j0 + kq * 16) * FF + f;
            __bf16 tmp[16];
            #pragma unroll
            for (int u = 0; u < 16; ++u) tmp[u] = (__bf16)vp[(size_t)u * FF];
            v8bf* dst = (v8bf*)&shV[f][kq * 16];
            dst[0] = *(v8bf*)&tmp[0];
            dst[1] = *(v8bf*)&tmp[8];
        }
        __syncthreads();

        // A fragment (strided global loads, f32 -> bf16), fold deg accumulation.
        v16bf afrag;
        #pragma unroll
        for (int e = 0; e < 16; ++e) {
            const int K = kloA + e + ((e >= 8) ? 8 : 0);
            const float a = aPtr[(size_t)(j0 + K) * LL];
            dsum += a;
            afrag[e] = (__bf16)a;
        }
        if (step + 1 < NSTEP)
            __builtin_prefetch(aPtr + (size_t)(j0 + KTILE) * LL, 0, 0); // global_prefetch_b8

        // 4 N-tiles of F: B frag from LDS (two ds_load_b128 each) + WMMA.
        #pragma unroll
        for (int t = 0; t < 4; ++t) {
            const int f = t * 16 + lm;
            v8bf lo = *(v8bf*)&shV[f][kloA];
            v8bf hi = *(v8bf*)&shV[f][kloA + 16];
            v16bf bfrag;
            #pragma unroll
            for (int e = 0; e < 8; ++e) { bfrag[e] = lo[e]; bfrag[e + 8] = hi[e]; }
            acc[t] = __builtin_amdgcn_wmma_f32_16x16x32_bf16(
                false, afrag, false, bfrag, (short)0, acc[t], false, false);
        }
        __syncthreads();
    }

    // ------------------------------ epilogue --------------------------------
    // deg[i, l]: lane & lane^16 hold complementary K halves of row lm.
    const float dall = dsum + __shfl_xor(dsum, 16, 32);
    if (lane < 16) degbuf[wave][lm] = dall;

    // agg -> shX[:, 320 + l*64 + f] as bf16 (C/D layout: row r / r+8 by lane half).
    #pragma unroll
    for (int t = 0; t < 4; ++t) {
        const int f = t * 16 + lm;
        #pragma unroll
        for (int r = 0; r < 8; ++r) {
            const int row = r + lhalf * 8;
            shX[row][FF + LL * FF + wave * FF + f] = (__bf16)acc[t][r];
        }
    }
    __syncthreads();

    // shX[:, 0:64] = V_i ; shX[:, 64 + l*64 + f] = deg[i,l] * V_i[f].
    {
        const int row = tid >> 3;            // 0..15
        const int f0  = (tid & 7) * 8;       // 0,8,...,56
        const float* vrow = V + (size_t)(b * NN + i0 + row) * FF + f0;
        float vv[8];
        #pragma unroll
        for (int u = 0; u < 8; ++u) vv[u] = vrow[u];
        __bf16 t0[8];
        #pragma unroll
        for (int u = 0; u < 8; ++u) t0[u] = (__bf16)vv[u];
        *(v8bf*)&shX[row][f0] = *(v8bf*)&t0[0];
        #pragma unroll
        for (int l = 0; l < LL; ++l) {
            const float d = degbuf[l][row];
            __bf16 t1[8];
            #pragma unroll
            for (int u = 0; u < 8; ++u) t1[u] = (__bf16)(d * vv[u]);
            *(v8bf*)&shX[row][FF + l * FF + f0] = *(v8bf*)&t1[0];
        }
    }
    __syncthreads();

    // out[:, wave*16 .. +15] = sigmoid( shX(16x576) @ [w1; w2; -w3](576x64) ).
    v8f oc = (v8f){0,0,0,0,0,0,0,0};
    const int cc = wave * 16 + lm;
    for (int kk = 0; kk < NKEP; ++kk) {
        const int kb = kk * KTILE + kloA;
        v8bf lo = *(v8bf*)&shX[lm][kb];
        v8bf hi = *(v8bf*)&shX[lm][kb + 16];
        v16bf afrag;
        #pragma unroll
        for (int e = 0; e < 8; ++e) { afrag[e] = lo[e]; afrag[e + 8] = hi[e]; }

        const int kg = kk * KTILE;
        const float* wp;
        float sgn = 1.0f;
        if (kg < FF)               { wp = w1 + (size_t)kg * CC; }
        else if (kg < FF + LL*FF)  { wp = w2 + (size_t)(kg - FF) * CC; }
        else                       { wp = w3 + (size_t)(kg - FF - LL*FF) * CC; sgn = -1.0f; }

        v16bf bfrag;
        #pragma unroll
        for (int e = 0; e < 16; ++e)
            bfrag[e] = (__bf16)(sgn * wp[(size_t)(kloA + e + ((e >= 8) ? 8 : 0)) * CC + cc]);

        oc = __builtin_amdgcn_wmma_f32_16x16x32_bf16(
            false, afrag, false, bfrag, (short)0, oc, false, false);
    }

    float* outp = out + (size_t)(b * NN + i0) * CC;
    #pragma unroll
    for (int r = 0; r < 8; ++r) {
        const int row = r + lhalf * 8;
        const float v = oc[r];
        outp[(size_t)row * CC + cc] = 1.0f / (1.0f + __expf(-v));
    }
}

extern "C" void kernel_launch(void* const* d_in, const int* in_sizes, int n_in,
                              void* d_out, int out_size, void* d_ws, size_t ws_size,
                              hipStream_t stream) {
    (void)in_sizes; (void)n_in; (void)d_ws; (void)ws_size; (void)out_size;
    const float* V  = (const float*)d_in[0];
    const float* A  = (const float*)d_in[1];
    const float* w1 = (const float*)d_in[2];
    const float* w2 = (const float*)d_in[3];
    const float* w3 = (const float*)d_in[4];
    float* out = (float*)d_out;

    dim3 grid(BB * (NN / 16));
    dim3 block(128);
    LEConvMultiEdge_87514253624018_kernel<<<grid, block, 0, stream>>>(V, A, w1, w2, w3, out);
}
